// FusedGWAttentionFF_46669114638476
// MI455X (gfx1250) — compile-verified, matrix operand order checked by reading, not run on hardware
//
#include <hip/hip_runtime.h>
#include <hip/hip_bf16.h>

// ---------------- types ----------------
typedef __bf16 bf16;
typedef __attribute__((ext_vector_type(16))) __bf16 v16bf;
typedef __attribute__((ext_vector_type(8)))  __bf16 v8bf;
typedef __attribute__((ext_vector_type(8)))  float  v8f;

union FragU { v16bf v; v8bf h[2]; };

__device__ __forceinline__ bf16 f2bf(float f) {
    union { float f; unsigned u; } a; a.f = f;
    unsigned r = a.u + 0x7FFFu + ((a.u >> 16) & 1u);
    union { unsigned short s; bf16 b; } o; o.s = (unsigned short)(r >> 16);
    return o.b;
}
__device__ __forceinline__ float bf2f(bf16 b) {
    union { unsigned short s; bf16 b; } i; i.b = b;
    union { unsigned u; float f; } o; o.u = ((unsigned)i.s) << 16;
    return o.f;
}

__device__ __forceinline__ v8f wmma_bf16(v16bf a, v16bf b, v8f c) {
    return __builtin_amdgcn_wmma_f32_16x16x32_bf16(
        /*neg_a=*/false, a, /*neg_b=*/false, b,
        /*c_mod=*/(short)0, c, /*reuse_a=*/false, /*reuse_b=*/false);
}

// A/B fragment, row-major source: lane (r = lane&15) reads 2x8 contiguous bf16.
// base points at (row0, k0) of the tile; 16B alignment guaranteed by layout.
__device__ __forceinline__ v16bf load_frag_row(const bf16* base, int stride, int lane) {
    int r  = lane & 15;
    int kh = lane >> 4;
    FragU u;
    u.h[0] = *(const v8bf*)(base + r * stride + kh * 8);
    u.h[1] = *(const v8bf*)(base + r * stride + 16 + kh * 8);
    return u.v;
}

// B fragment where K advances along rows of the source (strided per element).
// base points at (k0-row, n0-col).
__device__ __forceinline__ v16bf load_frag_col(const bf16* base, int stride, int lane) {
    int n  = lane & 15;
    int kh = lane >> 4;
    v16bf v;
#pragma unroll
    for (int j = 0; j < 8; ++j) {
        v[j]     = base[(kh * 8 + j) * stride + n];
        v[j + 8] = base[(16 + kh * 8 + j) * stride + n];
    }
    return v;
}

// ---------------- constants ----------------
#define DIMC   384
#define HEADS  12
#define DH     32
#define WIN    8
#define NTOK   64
#define DIMFF  1536

#define SXK    392                  // bf16 row stride for 64x384 tiles (pad 8)
#define SCK    72                   // score row stride (64 + 8)
#define SFK    264                  // ff chunk row stride (256 + 8)

#define X_OFF     0
#define Q_OFF     50176
#define KV_OFF    100352
#define SC_OFF    150528            // 12 heads x (64*72*2 = 9216 B)
#define FF_OFF    261120            // 64 x 264 bf16 = 33792 B
#define GATE_OFF  294912            // 64 x 16 f32  = 4096 B
#define SMEM_BYTES 299008

#define NW_FUSED  (2304 * 384)      // 884736
#define NW_ATTN   (384 * 384)       // 147456
#define NW_FFOUT  (384 * 1536)      // 589824
#define NW_TOTAL  (NW_FUSED + NW_ATTN + NW_FFOUT)

// ---------------- prep: fp32 -> bf16 weight conversion ----------------
__global__ __launch_bounds__(256) void convert_weights_kernel(
    const float* __restrict__ wf, const float* __restrict__ wa,
    const float* __restrict__ wo, bf16* __restrict__ dst) {
    int i = blockIdx.x * 256 + threadIdx.x;
    if (i < NW_FUSED)                    dst[i] = f2bf(wf[i]);
    else if (i < NW_FUSED + NW_ATTN)     dst[i] = f2bf(wa[i - NW_FUSED]);
    else if (i < NW_TOTAL)               dst[i] = f2bf(wo[i - NW_FUSED - NW_ATTN]);
}

// ---------------- fused window kernel ----------------
__global__ __launch_bounds__(256) void fused_gwattn_ff_kernel(
    const float* __restrict__ x,
    const float* __restrict__ w_gate,
    const float* __restrict__ b_gate,
    const float* __restrict__ sr_scale_p,
    const float* __restrict__ sr_bias_p,
    const bf16*  __restrict__ wfb,   // [2304,384] bf16
    const bf16*  __restrict__ wab,   // [384,384]  bf16
    const bf16*  __restrict__ wob,   // [384,1536] bf16
    float* __restrict__ out) {
    extern __shared__ char smem[];
    bf16*  sX    = (bf16*)(smem + X_OFF);
    bf16*  sQ    = (bf16*)(smem + Q_OFF);
    bf16*  sKV   = (bf16*)(smem + KV_OFF);
    bf16*  sSC   = (bf16*)(smem + SC_OFF);
    bf16*  sFF   = (bf16*)(smem + FF_OFF);
    float* sGate = (float*)(smem + GATE_OFF);
    float* sStage = (float*)(smem + 0);      // reused at the end (64 x 388 f32)

    const int tid  = threadIdx.x;
    const int lane = tid & 31;
    const int wave = tid >> 5;

    const int wi  = blockIdx.x;          // 0..1023
    const int b   = wi >> 6;
    const int rem = wi & 63;
    const int h0  = (rem >> 3) << 3;
    const int w0  = (rem & 7) << 3;

    const float srs = sr_scale_p[0];
    const float srb = sr_bias_p[0];

    // -------- phase 1: gather X window -> LDS bf16 [token][channel] --------
    const float* xwin = x + (size_t)b * DIMC * 4096 + h0 * 64 + w0;
    for (int idx = tid; idx < NTOK * DIMC; idx += 256) {
        int c = idx >> 6, t = idx & 63;
        int p1 = t >> 3, p2 = t & 7;
        sX[t * SXK + c] = f2bf(xwin[(size_t)c * 4096 + p1 * 64 + p2]);
    }
    __builtin_prefetch(wfb + (size_t)wave * 4096, 0, 1);   // global_prefetch_b8
    __syncthreads();

    // -------- phase 2: Q = X*Wq^T, KV = X*Wk^T (192 output tiles) --------
    for (int tile = wave; tile < 192; tile += 8) {
        int sel = tile >= 96;
        int tt  = tile - sel * 96;
        int mt  = tt / 24, ct = tt % 24;
        const bf16* wrow = wfb + (size_t)(sel * DIMC + ct * 16) * DIMC;
        v8f acc = {};
#pragma unroll
        for (int k0 = 0; k0 < DIMC; k0 += 32) {
            v16bf a  = load_frag_row(sX + mt * 16 * SXK + k0, SXK, lane);
            v16bf bb = load_frag_row(wrow + k0, DIMC, lane);
            acc = wmma_bf16(a, bb, acc);
        }
        bf16* dst = sel ? sKV : sQ;
        int row0 = mt * 16 + ((lane >> 4) << 3);
        int col  = ct * 16 + (lane & 15);
#pragma unroll
        for (int r = 0; r < 8; ++r) dst[(row0 + r) * SXK + col] = f2bf(acc[r]);
    }

    // -------- phase 3a: gate[t][h] = sigmoid(x.w_gate + b) (VALU) --------
    for (int idx = tid; idx < NTOK * HEADS; idx += 256) {
        int t = idx & 63, h = idx >> 6;
        const float* wg = w_gate + h * DIMC;
        float g = b_gate[h];
        for (int c = 0; c < DIMC; c += 4) {
            g += bf2f(sX[t * SXK + c])     * wg[c]
               + bf2f(sX[t * SXK + c + 1]) * wg[c + 1]
               + bf2f(sX[t * SXK + c + 2]) * wg[c + 2]
               + bf2f(sX[t * SXK + c + 3]) * wg[c + 3];
        }
        sGate[t * 16 + h] = 1.f / (1.f + __expf(-g));
    }
    __syncthreads();

    // -------- phase 3b: scores + softmax per head (in registers) --------
    const float scale = 0.17677669529663689f;  // 1/sqrt(32)
    for (int h = wave; h < HEADS; h += 8) {
        const int hoff = h * DH;
        float sc[4][4][8];
#pragma unroll
        for (int mt = 0; mt < 4; ++mt) {
            v16bf a = load_frag_row(sQ + mt * 16 * SXK + hoff, SXK, lane);
#pragma unroll
            for (int nt = 0; nt < 4; ++nt) {
                v16bf bb = load_frag_row(sKV + nt * 16 * SXK + hoff, SXK, lane);
                v8f acc = {};
                acc = wmma_bf16(a, bb, acc);
#pragma unroll
                for (int r = 0; r < 8; ++r) sc[mt][nt][r] = acc[r];
            }
        }
        bf16* sA = sSC + h * (NTOK * SCK);
#pragma unroll
        for (int mt = 0; mt < 4; ++mt) {
#pragma unroll
            for (int r = 0; r < 8; ++r) {
                // row = mt*16 + r (+8 for upper half); columns spread over 16 lanes x 4 nt
                float mx = fmaxf(fmaxf(sc[mt][0][r], sc[mt][1][r]),
                                 fmaxf(sc[mt][2][r], sc[mt][3][r]));
                mx = fmaxf(mx, __shfl_xor(mx, 1));
                mx = fmaxf(mx, __shfl_xor(mx, 2));
                mx = fmaxf(mx, __shfl_xor(mx, 4));
                mx = fmaxf(mx, __shfl_xor(mx, 8));
                float e[4], s = 0.f;
#pragma unroll
                for (int nt = 0; nt < 4; ++nt) {
                    e[nt] = __expf((sc[mt][nt][r] - mx) * scale);
                    s += e[nt];
                }
                s += __shfl_xor(s, 1);
                s += __shfl_xor(s, 2);
                s += __shfl_xor(s, 4);
                s += __shfl_xor(s, 8);
                float inv = 1.f / s;
                int row = mt * 16 + r + ((lane >> 4) << 3);
#pragma unroll
                for (int nt = 0; nt < 4; ++nt)
                    sA[row * SCK + nt * 16 + (lane & 15)] = f2bf(e[nt] * inv);
            }
        }
    }
    __syncthreads();

    // -------- phase 3c: x_attn = attn * KV, gated; write into sQ --------
    for (int h = wave; h < HEADS; h += 8) {
        const bf16* sA = sSC + h * (NTOK * SCK);
        const int hoff = h * DH;
#pragma unroll
        for (int mt = 0; mt < 4; ++mt) {
#pragma unroll
            for (int dt = 0; dt < 2; ++dt) {
                v8f acc = {};
#pragma unroll
                for (int k0 = 0; k0 < NTOK; k0 += 32) {
                    v16bf a  = load_frag_row(sA + mt * 16 * SCK + k0, SCK, lane);
                    v16bf bb = load_frag_col(sKV + k0 * SXK + hoff + dt * 16, SXK, lane);
                    acc = wmma_bf16(a, bb, acc);
                }
                int row0 = mt * 16 + ((lane >> 4) << 3);
                int col  = hoff + dt * 16 + (lane & 15);
#pragma unroll
                for (int r = 0; r < 8; ++r) {
                    float g = sGate[(row0 + r) * 16 + h];
                    sQ[(row0 + r) * SXK + col] = f2bf(acc[r] * g);
                }
            }
        }
    }
    __syncthreads();

    // -------- phase 5: out = x_attn*Wo^T + StarReLU(X*Wff^T)*Wffo^T --------
    v8f acc[12];
    int mts[12], ots[12];
#pragma unroll
    for (int i = 0; i < 12; ++i) {
        int tile = wave + i * 8;           // 96 tiles: (mt 0..3) x (ot 0..23)
        mts[i] = tile / 24;
        ots[i] = tile % 24;
        acc[i] = (v8f){};
    }

    // attention projection
    __builtin_prefetch(wab + (size_t)wave * 4096, 0, 1);
#pragma unroll
    for (int i = 0; i < 12; ++i) {
        const bf16* wrow = wab + (size_t)ots[i] * 16 * DIMC;
        for (int k0 = 0; k0 < DIMC; k0 += 32) {
            v16bf a  = load_frag_row(sQ + mts[i] * 16 * SXK + k0, SXK, lane);
            v16bf bb = load_frag_row(wrow + k0, DIMC, lane);
            acc[i] = wmma_bf16(a, bb, acc[i]);
        }
    }

    // FF path in 6 chunks of 256 columns
    for (int ch = 0; ch < 6; ++ch) {
        const int fbase = 2 * DIMC + ch * 256;   // rows 768.. of w_fused
        __builtin_prefetch(wfb + (size_t)fbase * DIMC, 0, 1);
        for (int t2 = wave; t2 < 64; t2 += 8) {  // (mt 0..3) x (ft 0..15)
            int mt = t2 >> 4, ft = t2 & 15;
            v8f fa = {};
#pragma unroll
            for (int k0 = 0; k0 < DIMC; k0 += 32) {
                v16bf a  = load_frag_row(sX + mt * 16 * SXK + k0, SXK, lane);
                v16bf bb = load_frag_row(wfb + (size_t)(fbase + ft * 16) * DIMC + k0,
                                         DIMC, lane);
                fa = wmma_bf16(a, bb, fa);
            }
            int row0 = mt * 16 + ((lane >> 4) << 3);
            int col  = ft * 16 + (lane & 15);
#pragma unroll
            for (int r = 0; r < 8; ++r) {
                float v = fa[r];
                v = v > 0.f ? v : 0.f;           // StarReLU
                v = srs * v * v + srb;
                sFF[(row0 + r) * SFK + col] = f2bf(v);
            }
        }
        __syncthreads();
#pragma unroll
        for (int i = 0; i < 12; ++i) {
            const bf16* wrow = wob + (size_t)ots[i] * 16 * DIMFF + ch * 256;
            for (int k0 = 0; k0 < 256; k0 += 32) {
                v16bf a  = load_frag_row(sFF + mts[i] * 16 * SFK + k0, SFK, lane);
                v16bf bb = load_frag_row(wrow + k0, DIMFF, lane);
                acc[i] = wmma_bf16(a, bb, acc[i]);
            }
        }
        __syncthreads();
    }

    // -------- phase 6: stage in LDS, coalesced scatter to NCHW --------
#pragma unroll
    for (int i = 0; i < 12; ++i) {
        int row0 = mts[i] * 16 + ((lane >> 4) << 3);
        int col  = ots[i] * 16 + (lane & 15);
#pragma unroll
        for (int r = 0; r < 8; ++r) sStage[(row0 + r) * 388 + col] = acc[i][r];
    }
    __syncthreads();
    float* owin = out + (size_t)b * DIMC * 4096 + h0 * 64 + w0;
    for (int idx = tid; idx < NTOK * DIMC; idx += 256) {
        int o = idx >> 6, t = idx & 63;
        int p1 = t >> 3, p2 = t & 7;
        owin[(size_t)o * 4096 + p1 * 64 + p2] = sStage[t * 388 + o];
    }
}

// ---------------- launch ----------------
extern "C" void kernel_launch(void* const* d_in, const int* in_sizes, int n_in,
                              void* d_out, int out_size, void* d_ws, size_t ws_size,
                              hipStream_t stream) {
    const float* x          = (const float*)d_in[0];
    const float* w_fused    = (const float*)d_in[1];
    const float* w_gate     = (const float*)d_in[2];
    const float* b_gate     = (const float*)d_in[3];
    const float* w_attn_out = (const float*)d_in[4];
    const float* w_ff_out   = (const float*)d_in[5];
    const float* sr_scale   = (const float*)d_in[6];
    const float* sr_bias    = (const float*)d_in[7];
    float* out = (float*)d_out;

    bf16* wsb = (bf16*)d_ws;
    bf16* wfb = wsb;                       // [2304,384]
    bf16* wab = wsb + NW_FUSED;            // [384,384]
    bf16* wob = wsb + NW_FUSED + NW_ATTN;  // [384,1536]

    convert_weights_kernel<<<(NW_TOTAL + 255) / 256, 256, 0, stream>>>(
        w_fused, w_attn_out, w_ff_out, wsb);

    hipFuncSetAttribute((const void*)fused_gwattn_ff_kernel,
                        hipFuncAttributeMaxDynamicSharedMemorySize, SMEM_BYTES);
    fused_gwattn_ff_kernel<<<1024, 256, SMEM_BYTES, stream>>>(
        x, w_gate, b_gate, sr_scale, sr_bias, wfb, wab, wob, out);
}
